// PointerGeneratorDecoder_67473936220543
// MI455X (gfx1250) — matrix operand order repaired and checked by reading, not compiled, not run on hardware
//
#include <hip/hip_runtime.h>
#include <hip/hip_bf16.h>
#include <math.h>

#define B_    8
#define TOUT  64
#define TSRC  512
#define VOCAB 50257
#define VPAD  50272   // 3142 * 16
#define EMB   256
#define ENC   512
#define H_    512

typedef __attribute__((ext_vector_type(16))) __bf16 v16bf;
typedef __attribute__((ext_vector_type(8)))  __bf16 v8bf;
typedef __attribute__((ext_vector_type(8)))  float  v8f;

__device__ __forceinline__ unsigned short f2bf(float f) {
    unsigned int u = __float_as_uint(f);
    u += 0x7FFFu + ((u >> 16) & 1u);            // round-to-nearest-even
    return (unsigned short)(u >> 16);
}

// A fragment: 16x32 bf16, row-major source, lda in elements.
// lanes 0-15 = rows M (lower K half), lanes 16-31 = same rows (upper K half).
// elements 0..7 -> K = k0 + half*8 + i ; elements 8..15 -> K = k0 + 16 + half*8 + i
__device__ __forceinline__ v16bf load_frag_a(const __bf16* A, int lda, int k0, int lane) {
    int m  = lane & 15;
    int hf = lane >> 4;
    const __bf16* p = A + (size_t)m * lda + k0 + hf * 8;
    v8bf lo = *(const v8bf*)(p);
    v8bf hi = *(const v8bf*)(p + 16);
    v16bf r;
#pragma unroll
    for (int i = 0; i < 8; ++i) { r[i] = lo[i]; r[i + 8] = hi[i]; }
    return r;
}

// B fragment: 32x16 bf16; column n contiguous in memory (row n of weight matrix).
// lane = column; elements 0..7 -> K = k0 + half*8 + i ; elements 8..15 -> K = k0 + 16 + half*8 + i
__device__ __forceinline__ v16bf load_frag_b(const __bf16* Bm, int ldb, int k0, int lane) {
    int n  = lane & 15;
    int hf = lane >> 4;
    const __bf16* p = Bm + (size_t)n * ldb + k0 + hf * 8;
    v8bf lo = *(const v8bf*)(p);
    v8bf hi = *(const v8bf*)(p + 16);
    v16bf r;
#pragma unroll
    for (int i = 0; i < 8; ++i) { r[i] = lo[i]; r[i + 8] = hi[i]; }
    return r;
}

__device__ __forceinline__ float sigmoidf_(float x) { return 1.f / (1.f + __expf(-x)); }

// ---------------- conversion / init ----------------
__global__ void k_f32_to_bf16(const float* __restrict__ in, unsigned short* __restrict__ out,
                              int n_src, int n_total) {
    int i = blockIdx.x * blockDim.x + threadIdx.x;
    if (i >= n_total) return;
    out[i] = (i < n_src) ? f2bf(in[i]) : (unsigned short)0;
}

__global__ void k_init_state(const float* __restrict__ h0, const float* __restrict__ c0,
                             float* __restrict__ h_cur, float* __restrict__ c_cur) {
    int i = blockIdx.x * blockDim.x + threadIdx.x;
    if (i < B_ * H_) { h_cur[i] = h0[i]; c_cur[i] = c0[i]; }
}

// ---------------- encp = enc_outputs @ enc_proj_w.T + b (WMMA bf16) ----------------
__global__ void k_encp_gemm(const unsigned short* __restrict__ encO_bf,
                            const unsigned short* __restrict__ Wp_bf,
                            const float* __restrict__ bias, float* __restrict__ encp) {
    int wave = (blockIdx.x * blockDim.x + threadIdx.x) >> 5;
    int lane = threadIdx.x & 31;
    int m_tile = wave >> 5;     // 32 n-tiles (H/16)
    int n_tile = wave & 31;
    const __bf16* A  = (const __bf16*)encO_bf + (size_t)m_tile * 16 * ENC;
    const __bf16* Bm = (const __bf16*)Wp_bf   + (size_t)n_tile * 16 * ENC;
    v8f acc = {};
    for (int k0 = 0; k0 < ENC; k0 += 32) {
        v16bf a = load_frag_a(A, ENC, k0, lane);
        v16bf b = load_frag_b(Bm, ENC, k0, lane);
        acc = __builtin_amdgcn_wmma_f32_16x16x32_bf16(false, a, false, b, (short)0, acc, false, false);
    }
    int n  = n_tile * 16 + (lane & 15);
    int mb = m_tile * 16 + ((lane >> 4) ? 8 : 0);
    float bv = bias[n];
#pragma unroll
    for (int i = 0; i < 8; ++i) encp[(size_t)(mb + i) * H_ + n] = acc[i] + bv;
}

// ---------------- logits = h16 @ out_w_bf.T + out_b (WMMA bf16), 2 N-tiles / wave ----------------
__global__ void k_vocab_gemm(const unsigned short* __restrict__ h16,
                             const unsigned short* __restrict__ Wv_bf,
                             const float* __restrict__ out_b, float* __restrict__ logits) {
    int wave = (blockIdx.x * blockDim.x + threadIdx.x) >> 5;
    int lane = threadIdx.x & 31;
    int nt0  = wave * 2;                           // two adjacent 16-wide N-tiles per wave
    if (nt0 * 16 >= VPAD) return;                  // wave-uniform: EXEC all-ones for WMMA
    const __bf16* A  = (const __bf16*)h16;
    const __bf16* B0 = (const __bf16*)Wv_bf + (size_t)nt0 * 16 * H_;
    const __bf16* B1 = B0 + (size_t)16 * H_;
    v8f acc0 = {}, acc1 = {};
    for (int k0 = 0; k0 < H_; k0 += 32) {
        v16bf a  = load_frag_a(A, H_, k0, lane);   // shared A fragment
        v16bf b0 = load_frag_b(B0, H_, k0, lane);
        v16bf b1 = load_frag_b(B1, H_, k0, lane);
        acc0 = __builtin_amdgcn_wmma_f32_16x16x32_bf16(false, a, false, b0, (short)0, acc0, false, false);
        acc1 = __builtin_amdgcn_wmma_f32_16x16x32_bf16(false, a, false, b1, (short)0, acc1, false, false);
    }
    if (lane < 16) {                               // rows M=0..7 live in VGPR i, lanes 0-15
        int n0 = nt0 * 16 + lane;
        int n1 = n0 + 16;
        float bv0 = (n0 < VOCAB) ? out_b[n0] : 0.f;
        float bv1 = (n1 < VOCAB) ? out_b[n1] : 0.f;
#pragma unroll
        for (int i = 0; i < 8; ++i) {
            logits[(size_t)i * VPAD + n0] = acc0[i] + bv0;
            logits[(size_t)i * VPAD + n1] = acc1[i] + bv1;
        }
    }
}

// ---------------- attention: scores, masked softmax, context ----------------
__global__ void k_attn(const float* __restrict__ h_cur, const float* __restrict__ encp,
                       const int* __restrict__ src_ids, float* __restrict__ attn_out,
                       float* __restrict__ context, int t) {
    int b = blockIdx.x;
    int tid = threadIdx.x;                         // 256 threads
    __shared__ float hs[H_];
    __shared__ float sc[TSRC];
    __shared__ float red[256];
    hs[tid]       = h_cur[b * H_ + tid];
    hs[tid + 256] = h_cur[b * H_ + tid + 256];
    __syncthreads();
    float lmax = -INFINITY;
    for (int s = tid; s < TSRC; s += 256) {
        const float4* row4 = (const float4*)(encp + ((size_t)(b * TSRC + s)) * H_);
        float d = 0.f;
#pragma unroll 4
        for (int k = 0; k < H_ / 4; ++k) {
            float4 v = row4[k];
            d += hs[4 * k] * v.x + hs[4 * k + 1] * v.y + hs[4 * k + 2] * v.z + hs[4 * k + 3] * v.w;
        }
        if (src_ids[b * TSRC + s] == 1) d = -INFINITY;
        sc[s] = d;
        lmax = fmaxf(lmax, d);
    }
    red[tid] = lmax; __syncthreads();
    for (int o = 128; o > 0; o >>= 1) { if (tid < o) red[tid] = fmaxf(red[tid], red[tid + o]); __syncthreads(); }
    float mx = red[0]; __syncthreads();
    float lsum = 0.f;
    for (int s = tid; s < TSRC; s += 256) { float e = __expf(sc[s] - mx); sc[s] = e; lsum += e; }
    red[tid] = lsum; __syncthreads();
    for (int o = 128; o > 0; o >>= 1) { if (tid < o) red[tid] += red[tid + o]; __syncthreads(); }
    float inv = 1.f / red[0]; __syncthreads();
    for (int s = tid; s < TSRC; s += 256) {
        float a = sc[s] * inv;
        sc[s] = a;
        attn_out[((size_t)(b * TOUT + t)) * TSRC + s] = a;
    }
    __syncthreads();
    for (int hc = tid; hc < H_; hc += 256) {       // lanes coalesced across hc at each s
        float acc = 0.f;
        for (int s = 0; s < TSRC; ++s) acc += sc[s] * encp[((size_t)(b * TSRC + s)) * H_ + hc];
        context[b * H_ + hc] = acc;
    }
}

// ---------------- LSTM gates ----------------
__global__ void k_gates(const float* __restrict__ h_cur, const float* __restrict__ context,
                        const int* __restrict__ tgt_ids, const float* __restrict__ embed,
                        const float* __restrict__ w_ih, const float* __restrict__ w_hh,
                        const float* __restrict__ b_ih, const float* __restrict__ b_hh,
                        float* __restrict__ gates, int t) {
    int idx = blockIdx.x * blockDim.x + threadIdx.x;
    if (idx >= B_ * 4 * H_) return;
    int b = idx >> 11;
    int r = idx & (4 * H_ - 1);
    float acc = b_ih[r] + b_hh[r];
    const float* wih = w_ih + (size_t)r * (EMB + H_);
    const float* emb = embed + (size_t)tgt_ids[b * TOUT + t] * EMB;
    for (int k = 0; k < EMB; ++k) acc += wih[k] * emb[k];
    const float* ctx = context + b * H_;
    for (int k = 0; k < H_; ++k) acc += wih[EMB + k] * ctx[k];
    const float* whh = w_hh + (size_t)r * H_;
    const float* hb  = h_cur + b * H_;
    for (int k = 0; k < H_; ++k) acc += whh[k] * hb[k];
    gates[b * 4 * H_ + r] = acc;
}

// ---------------- cell/state update + bf16 A-matrix (rows 8..15 zero) ----------------
__global__ void k_combine(const float* __restrict__ gates, float* __restrict__ h_cur,
                          float* __restrict__ c_cur, unsigned short* __restrict__ h16,
                          float* __restrict__ h_out, float* __restrict__ c_out) {
    int idx = blockIdx.x * blockDim.x + threadIdx.x;    // 16*512
    if (idx >= 16 * H_) return;
    int m  = idx >> 9;
    int hc = idx & (H_ - 1);
    if (m < B_) {
        const float* g = gates + m * 4 * H_;
        float gi = g[hc], gf = g[H_ + hc], gg = g[2 * H_ + hc], go = g[3 * H_ + hc];
        float c_new = sigmoidf_(gf) * c_cur[m * H_ + hc] + sigmoidf_(gi) * tanhf(gg);
        float h_new = sigmoidf_(go) * tanhf(c_new);
        c_cur[m * H_ + hc] = c_new;
        h_cur[m * H_ + hc] = h_new;
        h_out[m * H_ + hc] = h_new;
        c_out[m * H_ + hc] = c_new;
        h16[m * H_ + hc] = f2bf(h_new);
    } else {
        h16[m * H_ + hc] = 0;
    }
}

// ---------------- p_gen ----------------
__global__ void k_pgen(const float* __restrict__ h_cur, const float* __restrict__ context,
                       const int* __restrict__ tgt_ids, const float* __restrict__ embed,
                       const float* __restrict__ w_gen_w, const float* __restrict__ w_gen_b,
                       float* __restrict__ pgen, int t) {
    int b = blockIdx.x;
    int tid = threadIdx.x;                          // 256
    __shared__ float red[256];
    const float* emb = embed + (size_t)tgt_ids[b * TOUT + t] * EMB;
    float acc = 0.f;
    for (int j = tid; j < 2 * H_ + EMB; j += 256) {
        float x = (j < H_) ? h_cur[b * H_ + j]
                : (j < 2 * H_) ? context[b * H_ + (j - H_)]
                : emb[j - 2 * H_];
        acc += x * w_gen_w[j];
    }
    red[tid] = acc; __syncthreads();
    for (int o = 128; o > 0; o >>= 1) { if (tid < o) red[tid] += red[tid + o]; __syncthreads(); }
    if (tid == 0) pgen[b] = sigmoidf_(red[0] + w_gen_b[0]);
}

// ---------------- vocab softmax statistics ----------------
__global__ void k_smax_stats(const float* __restrict__ logits, float* __restrict__ stats) {
    int b = blockIdx.x;
    int tid = threadIdx.x;                          // 1024
    __shared__ float red[1024];
    float lm = -INFINITY;
    for (int v = tid; v < VOCAB; v += 1024) lm = fmaxf(lm, logits[(size_t)b * VPAD + v]);
    red[tid] = lm; __syncthreads();
    for (int o = 512; o > 0; o >>= 1) { if (tid < o) red[tid] = fmaxf(red[tid], red[tid + o]); __syncthreads(); }
    float mx = red[0]; __syncthreads();
    float ls = 0.f;
    for (int v = tid; v < VOCAB; v += 1024) ls += __expf(logits[(size_t)b * VPAD + v] - mx);
    red[tid] = ls; __syncthreads();
    for (int o = 512; o > 0; o >>= 1) { if (tid < o) red[tid] += red[tid + o]; __syncthreads(); }
    if (tid == 0) { stats[b] = mx; stats[8 + b] = red[0]; }
}

// ---------------- final mix: p_gen * softmax(logits) ----------------
__global__ void k_final(const float* __restrict__ logits, const float* __restrict__ stats,
                        const float* __restrict__ pgen, float* __restrict__ outs, int t) {
    size_t idx = (size_t)blockIdx.x * blockDim.x + threadIdx.x;
    if (idx >= (size_t)B_ * VOCAB) return;
    int b = (int)(idx / VOCAB);
    int v = (int)(idx % VOCAB);
    float val = pgen[b] * __expf(logits[(size_t)b * VPAD + v] - stats[b]) / stats[8 + b];
    outs[((size_t)(b * TOUT + t)) * VOCAB + v] = val;
}

// ---------------- copy-distribution scatter-add ----------------
__global__ void k_scatter(const float* __restrict__ attn_all, const int* __restrict__ src_ids,
                          const float* __restrict__ pgen, float* __restrict__ outs, int t) {
    int idx = blockIdx.x * blockDim.x + threadIdx.x;
    if (idx >= B_ * TSRC) return;
    int b = idx >> 9;
    int s = idx & (TSRC - 1);
    float a = attn_all[((size_t)(b * TOUT + t)) * TSRC + s];
    float w = (1.f - pgen[b]) * a;
    int tgt = src_ids[b * TSRC + s];
    atomicAdd(&outs[((size_t)(b * TOUT + t)) * VOCAB + tgt], w);
}

extern "C" void kernel_launch(void* const* d_in, const int* in_sizes, int n_in,
                              void* d_out, int out_size, void* d_ws, size_t ws_size,
                              hipStream_t stream) {
    (void)in_sizes; (void)n_in; (void)out_size; (void)ws_size;
    const int*   tgt_ids     = (const int*)d_in[0];
    const int*   src_ids     = (const int*)d_in[1];
    const float* enc_outputs = (const float*)d_in[2];
    const float* h0          = (const float*)d_in[3];
    const float* c0          = (const float*)d_in[4];
    const float* embed       = (const float*)d_in[5];
    const float* enc_proj_w  = (const float*)d_in[6];
    const float* enc_proj_b  = (const float*)d_in[7];
    const float* w_ih        = (const float*)d_in[8];
    const float* w_hh        = (const float*)d_in[9];
    const float* b_ih        = (const float*)d_in[10];
    const float* b_hh        = (const float*)d_in[11];
    const float* w_gen_w     = (const float*)d_in[12];
    const float* w_gen_b     = (const float*)d_in[13];
    const float* out_w       = (const float*)d_in[14];
    const float* out_b       = (const float*)d_in[15];

    float* outs     = (float*)d_out;
    float* h_out    = outs + (size_t)B_ * TOUT * VOCAB;
    float* c_out    = h_out + B_ * H_;
    float* attn_all = c_out + B_ * H_;

    char* ws = (char*)d_ws;
    size_t off = 0;
    auto carve = [&](size_t bytes) -> char* {
        char* p = ws + off;
        off = (off + bytes + 255) & ~(size_t)255;
        return p;
    };
    unsigned short* Wv_bf   = (unsigned short*)carve((size_t)VPAD * H_ * 2);   // 51.5 MB, L2-resident
    unsigned short* encO_bf = (unsigned short*)carve((size_t)B_ * TSRC * ENC * 2);
    unsigned short* Wp_bf   = (unsigned short*)carve((size_t)H_ * ENC * 2);
    float* encp    = (float*)carve((size_t)B_ * TSRC * H_ * 4);
    float* logits  = (float*)carve((size_t)B_ * VPAD * 4);
    float* h_cur   = (float*)carve(B_ * H_ * 4);
    float* c_cur   = (float*)carve(B_ * H_ * 4);
    unsigned short* h16 = (unsigned short*)carve(16 * H_ * 2);
    float* context = (float*)carve(B_ * H_ * 4);
    float* gates   = (float*)carve(B_ * 4 * H_ * 4);
    float* pgen    = (float*)carve(256);
    float* stats   = (float*)carve(256);

    // One-time (per call) conversions + encoder projection
    { int n = VPAD * H_;       k_f32_to_bf16<<<(n + 255) / 256, 256, 0, stream>>>(out_w, Wv_bf, VOCAB * H_, n); }
    { int n = B_ * TSRC * ENC; k_f32_to_bf16<<<(n + 255) / 256, 256, 0, stream>>>(enc_outputs, encO_bf, n, n); }
    { int n = H_ * ENC;        k_f32_to_bf16<<<(n + 255) / 256, 256, 0, stream>>>(enc_proj_w, Wp_bf, n, n); }
    k_init_state<<<(B_ * H_ + 255) / 256, 256, 0, stream>>>(h0, c0, h_cur, c_cur);
    k_encp_gemm<<<(B_ * TSRC / 16) * (H_ / 16) / 4, 128, 0, stream>>>(encO_bf, Wp_bf, enc_proj_b, encp);

    const int npairs = VPAD / 32;    // 1571 waves per step, 2 N-tiles each
    for (int t = 0; t < TOUT; ++t) {
        k_attn   <<<B_, 256, 0, stream>>>(h_cur, encp, src_ids, attn_all, context, t);
        k_gates  <<<(B_ * 4 * H_) / 256, 256, 0, stream>>>(h_cur, context, tgt_ids, embed,
                                                           w_ih, w_hh, b_ih, b_hh, gates, t);
        k_combine<<<(16 * H_) / 256, 256, 0, stream>>>(gates, h_cur, c_cur, h16, h_out, c_out);
        k_pgen   <<<B_, 256, 0, stream>>>(h_cur, context, tgt_ids, embed, w_gen_w, w_gen_b, pgen, t);
        k_vocab_gemm<<<(npairs + 3) / 4, 128, 0, stream>>>(h16, Wv_bf, out_b, logits);
        k_smax_stats<<<B_, 1024, 0, stream>>>(logits, stats);
        { size_t n = (size_t)B_ * VOCAB;
          k_final<<<(unsigned)((n + 255) / 256), 256, 0, stream>>>(logits, stats, pgen, outs, t); }
        k_scatter<<<(B_ * TSRC) / 256, 256, 0, stream>>>(attn_all, src_ids, pgen, outs, t);
    }
}
